// GRU_5050881540410
// MI455X (gfx1250) — compile-verified, matrix operand order checked by reading, not run on hardware
//
#include <hip/hip_runtime.h>
#include <hip/hip_bf16.h>

// GRU cell on MI455X (gfx1250), bf16 WMMA path.
//   z  = sigmoid(x@Wz.T + bwz + h@Uz.T + buz + b_z)
//   r  = sigmoid(x@Wr.T + bwr + h@Ur.T + bur + b_r)
//   hh = tanh   (x@Wh.T + bwh + (r*h)@Uh.T + buh + b_h)
//   out = (1-z)*h + z*hh         (h is a single broadcast row [1,O])
//
// Pipeline:
//   K0: fp32 -> bf16 converts (x, Wz, Wr, Wh, Uh)
//   K1: tiny GEMVs  uz = h@Uz.T + buz,  ur = h@Ur.T + bur   (h is 1 row)
//   K2: fused 3-gate WMMA GEMM over x; A tile double-buffered in LDS via
//       GLOBAL_LOAD_ASYNC_TO_LDS_B128 (ASYNCcnt) when available.
//       epilogue: z -> zbuf, a = r*h -> abuf(bf16), th = Gh+biases -> thbuf
//   K3: WMMA GEMM th += a@Uh.T, epilogue out = (1-z)*h + z*tanh(th)

#define BDIM 4096
#define IDIM 2048
#define ODIM 2048

typedef __attribute__((ext_vector_type(16))) __bf16 v16bf;
typedef __attribute__((ext_vector_type(8)))  __bf16 v8bf;
typedef __attribute__((ext_vector_type(4)))  __bf16 v4bf;
typedef __attribute__((ext_vector_type(8)))  float  v8f;
typedef __attribute__((ext_vector_type(4)))  float  v4f;

#define WMMA_BF16(A, Bm, C) \
  __builtin_amdgcn_wmma_f32_16x16x32_bf16(false, (A), false, (Bm), (short)0, (C), false, false)

// ---- CDNA5 async memory->LDS path (guarded; falls back to load+ds_store) ----
#if defined(__has_builtin)
#if __has_builtin(__builtin_amdgcn_global_load_async_to_lds_b128) && \
    __has_builtin(__builtin_amdgcn_s_wait_asynccnt)
#define GRU_HAS_ASYNC_LDS 1
#endif
#endif
#ifndef GRU_HAS_ASYNC_LDS
#define GRU_HAS_ASYNC_LDS 0
#endif

#if GRU_HAS_ASYNC_LDS
// Builtin signature (from compiler diagnostic): parameters are
//   as(1) v4i*  (printed as "vector_size(16) int __device__ *"),
//   as(3) v4i*, imm int offset, imm int cpol.
typedef int v4i_gl __attribute__((vector_size(16)));
typedef __attribute__((address_space(1))) v4i_gl as1_v4i;
typedef __attribute__((address_space(3))) v4i_gl as3_v4i;

__device__ __forceinline__ void gru_async_cp16(const __bf16* g, __bf16* l) {
  __builtin_amdgcn_global_load_async_to_lds_b128(
      (as1_v4i*)(void*)g, (as3_v4i*)(void*)l, 0, 0);
}
#endif

// ---------------------------------------------------------------- K0: convert
__global__ __launch_bounds__(256) void gru_cvt_bf16(const v4f* __restrict__ in,
                                                    v4bf* __restrict__ out, int n4) {
  int i = blockIdx.x * blockDim.x + threadIdx.x;
  if (i < n4) {
    v4f a = in[i];
    v4bf b;
    b[0] = (__bf16)a[0]; b[1] = (__bf16)a[1];
    b[2] = (__bf16)a[2]; b[3] = (__bf16)a[3];
    out[i] = b;
  }
}

// ---------------------------------------------------------------- K1: GEMVs
__global__ __launch_bounds__(256) void gru_gemv(const float* __restrict__ h,
                                                const float* __restrict__ Uz,
                                                const float* __restrict__ buz,
                                                const float* __restrict__ Ur,
                                                const float* __restrict__ bur,
                                                float* __restrict__ uzv,
                                                float* __restrict__ urv) {
  int t = blockIdx.x * blockDim.x + threadIdx.x;   // 0 .. 2*ODIM-1
  const float* Umat;
  const float* bias;
  float* outv;
  int n;
  if (t < ODIM) { Umat = Uz; bias = buz; outv = uzv; n = t; }
  else          { Umat = Ur; bias = bur; outv = urv; n = t - ODIM; }
  const float* row = Umat + (size_t)n * ODIM;
  float acc = 0.f;
#pragma unroll 4
  for (int k = 0; k < ODIM; ++k) acc += row[k] * h[k];
  outv[n] = acc + bias[n];
}

// -------------------------------------------------- K2: fused 3-gate WMMA GEMM
// Block: 256 thr = 8 waves, tile BM=64 x BN=64. Wave (wm,wn): wm in {0,1} rows,
// wn in {0..3} cols; wave computes 32x16 per gate (2 M-subtiles, 3 gates).
__global__ __launch_bounds__(256) void gru_gates_wmma(
    const __bf16* __restrict__ xb,   // [B,I]
    const __bf16* __restrict__ wz,   // [O,I]
    const __bf16* __restrict__ wr,   // [O,I]
    const __bf16* __restrict__ wh,   // [O,I]
    const float* __restrict__ bwz, const float* __restrict__ bwr,
    const float* __restrict__ bwh,
    const float* __restrict__ uzv, const float* __restrict__ urv,
    const float* __restrict__ buh,
    const float* __restrict__ b_z, const float* __restrict__ b_r,
    const float* __restrict__ b_h,
    const float* __restrict__ hvec,
    float* __restrict__ zbuf, float* __restrict__ thbuf,
    __bf16* __restrict__ abuf) {
  __shared__ __bf16 As[2][64 * 40];  // double-buffered 64x32 tile, stride 40

  const int tid  = threadIdx.x;
  const int lane = tid & 31;
  const int wid  = tid >> 5;
  const int wm   = wid & 1;
  const int wn   = wid >> 1;
  const int mBlock = blockIdx.y * 64;
  const int nWave  = blockIdx.x * 64 + wn * 16;
  const int l15 = lane & 15;         // A: row-in-tile / B: col-in-tile
  const int hi  = lane >> 4;         // half-wave selector (K interleave)

  v8f accz0 = {}, accz1 = {}, accr0 = {}, accr1 = {}, acch0 = {}, acch1 = {};

  // cooperative A-tile staging: 64 rows x 32 bf16; one 16B chunk per thread
  const int srow   = tid >> 2;
  const int schunk = (tid & 3) * 8;
  const int sdst   = srow * 40 + schunk;
  const size_t gArow = (size_t)(mBlock + srow) * IDIM;
  // B fragment base: weights row-major -> per-lane contiguous 16B loads
  const size_t bro = (size_t)(nWave + l15) * IDIM;

  const int abase0 = (wm * 32 + l15) * 40 + hi * 8;
  const int abase1 = abase0 + 16 * 40;
  const int NK = IDIM / 32;

#if GRU_HAS_ASYNC_LDS
  gru_async_cp16(&xb[gArow + schunk], &As[0][sdst]);
#endif

  for (int it = 0; it < NK; ++it) {
    const int k0 = it * 32;
#if GRU_HAS_ASYNC_LDS
    const int cur = it & 1;
    __builtin_amdgcn_s_wait_asynccnt(0);  // my slice of As[cur] has landed
    __syncthreads();                      // all slices landed; As[cur^1] free
    if (it + 1 < NK)
      gru_async_cp16(&xb[gArow + k0 + 32 + schunk], &As[cur ^ 1][sdst]);
    const __bf16* S = &As[cur][0];
#else
    __syncthreads();
    *(v8bf*)&As[0][sdst] = *(const v8bf*)&xb[gArow + k0 + schunk];
    __syncthreads();
    const __bf16* S = &As[0][0];
#endif

    union { v16bf v; v8bf h8[2]; } u;
    u.h8[0] = *(const v8bf*)&S[abase0];
    u.h8[1] = *(const v8bf*)&S[abase0 + 16];
    v16bf a0 = u.v;
    u.h8[0] = *(const v8bf*)&S[abase1];
    u.h8[1] = *(const v8bf*)&S[abase1 + 16];
    v16bf a1 = u.v;

    const size_t bo = bro + k0 + hi * 8;
    u.h8[0] = *(const v8bf*)&wz[bo];
    u.h8[1] = *(const v8bf*)&wz[bo + 16];
    v16bf bz = u.v;
    u.h8[0] = *(const v8bf*)&wr[bo];
    u.h8[1] = *(const v8bf*)&wr[bo + 16];
    v16bf br = u.v;
    u.h8[0] = *(const v8bf*)&wh[bo];
    u.h8[1] = *(const v8bf*)&wh[bo + 16];
    v16bf bh = u.v;

    accz0 = WMMA_BF16(a0, bz, accz0);
    accz1 = WMMA_BF16(a1, bz, accz1);
    accr0 = WMMA_BF16(a0, br, accr0);
    accr1 = WMMA_BF16(a1, br, accr1);
    acch0 = WMMA_BF16(a0, bh, acch0);
    acch1 = WMMA_BF16(a1, bh, acch1);
  }

  // epilogue: C/D layout -> VGPR j holds row j (+8 for lanes 16-31), col=lane&15
  const int col = nWave + l15;
  const float cz = bwz[col] + uzv[col] + b_z[col];
  const float cr = bwr[col] + urv[col] + b_r[col];
  const float ch = bwh[col] + buh[col] + b_h[col];
  const float hcol = hvec[col];
#pragma unroll
  for (int sub = 0; sub < 2; ++sub) {
    const v8f az = sub ? accz1 : accz0;
    const v8f ar = sub ? accr1 : accr0;
    const v8f ah = sub ? acch1 : acch0;
    const int rbase = mBlock + wm * 32 + sub * 16 + hi * 8;
#pragma unroll
    for (int j = 0; j < 8; ++j) {
      const size_t o = (size_t)(rbase + j) * ODIM + col;
      const float z = 1.0f / (1.0f + __expf(-(az[j] + cz)));
      const float r = 1.0f / (1.0f + __expf(-(ar[j] + cr)));
      zbuf[o]  = z;
      abuf[o]  = (__bf16)(r * hcol);
      thbuf[o] = ah[j] + ch;
    }
  }
}

// --------------------------------------- K3: th += a@Uh.T ; out = blend(tanh)
// Block: 8 waves, BM=64 x BN=128; wave computes 32x32 (2 M x 2 N subtiles).
__global__ __launch_bounds__(256) void gru_out_wmma(
    const __bf16* __restrict__ ab,   // [B,O]  r*h in bf16
    const __bf16* __restrict__ uh,   // [O,O]
    const float* __restrict__ thbuf,
    const float* __restrict__ zbuf,
    const float* __restrict__ hvec,
    float* __restrict__ out) {
  __shared__ __bf16 As[2][64 * 40];

  const int tid  = threadIdx.x;
  const int lane = tid & 31;
  const int wid  = tid >> 5;
  const int wm   = wid & 1;
  const int wn   = wid >> 1;
  const int mBlock = blockIdx.y * 64;
  const int nWave  = blockIdx.x * 128 + wn * 32;
  const int l15 = lane & 15;
  const int hi  = lane >> 4;

  v8f acc00 = {}, acc01 = {}, acc10 = {}, acc11 = {};

  const int srow   = tid >> 2;
  const int schunk = (tid & 3) * 8;
  const int sdst   = srow * 40 + schunk;
  const size_t gArow = (size_t)(mBlock + srow) * ODIM;
  const size_t bro0 = (size_t)(nWave + l15) * ODIM;
  const size_t bro1 = (size_t)(nWave + 16 + l15) * ODIM;

  const int abase0 = (wm * 32 + l15) * 40 + hi * 8;
  const int abase1 = abase0 + 16 * 40;
  const int NK = ODIM / 32;

#if GRU_HAS_ASYNC_LDS
  gru_async_cp16(&ab[gArow + schunk], &As[0][sdst]);
#endif

  for (int it = 0; it < NK; ++it) {
    const int k0 = it * 32;
#if GRU_HAS_ASYNC_LDS
    const int cur = it & 1;
    __builtin_amdgcn_s_wait_asynccnt(0);
    __syncthreads();
    if (it + 1 < NK)
      gru_async_cp16(&ab[gArow + k0 + 32 + schunk], &As[cur ^ 1][sdst]);
    const __bf16* S = &As[cur][0];
#else
    __syncthreads();
    *(v8bf*)&As[0][sdst] = *(const v8bf*)&ab[gArow + k0 + schunk];
    __syncthreads();
    const __bf16* S = &As[0][0];
#endif

    union { v16bf v; v8bf h8[2]; } u;
    u.h8[0] = *(const v8bf*)&S[abase0];
    u.h8[1] = *(const v8bf*)&S[abase0 + 16];
    v16bf a0 = u.v;
    u.h8[0] = *(const v8bf*)&S[abase1];
    u.h8[1] = *(const v8bf*)&S[abase1 + 16];
    v16bf a1 = u.v;

    u.h8[0] = *(const v8bf*)&uh[bro0 + k0 + hi * 8];
    u.h8[1] = *(const v8bf*)&uh[bro0 + k0 + hi * 8 + 16];
    v16bf b0 = u.v;
    u.h8[0] = *(const v8bf*)&uh[bro1 + k0 + hi * 8];
    u.h8[1] = *(const v8bf*)&uh[bro1 + k0 + hi * 8 + 16];
    v16bf b1 = u.v;

    acc00 = WMMA_BF16(a0, b0, acc00);
    acc01 = WMMA_BF16(a0, b1, acc01);
    acc10 = WMMA_BF16(a1, b0, acc10);
    acc11 = WMMA_BF16(a1, b1, acc11);
  }

#pragma unroll
  for (int ms = 0; ms < 2; ++ms) {
#pragma unroll
    for (int ns = 0; ns < 2; ++ns) {
      const v8f a = ms ? (ns ? acc11 : acc10) : (ns ? acc01 : acc00);
      const int rbase = mBlock + wm * 32 + ms * 16 + hi * 8;
      const int col   = nWave + ns * 16 + l15;
      const float hcol = hvec[col];
#pragma unroll
      for (int j = 0; j < 8; ++j) {
        const size_t o = (size_t)(rbase + j) * ODIM + col;
        const float th = a[j] + thbuf[o];
        const float hh = tanhf(th);
        const float z  = zbuf[o];
        out[o] = (1.0f - z) * hcol + z * hh;
      }
    }
  }
}

// ---------------------------------------------------------------- launch glue
static constexpr size_t XB_OFF  = 0;                                        // x bf16
static constexpr size_t WZB_OFF = XB_OFF  + (size_t)BDIM * IDIM * 2;        // Wz bf16
static constexpr size_t WRB_OFF = WZB_OFF + (size_t)ODIM * IDIM * 2;
static constexpr size_t WHB_OFF = WRB_OFF + (size_t)ODIM * IDIM * 2;
static constexpr size_t UHB_OFF = WHB_OFF + (size_t)ODIM * IDIM * 2;
static constexpr size_t ZB_OFF  = UHB_OFF + (size_t)ODIM * ODIM * 2;        // z fp32
static constexpr size_t THB_OFF = ZB_OFF  + (size_t)BDIM * ODIM * 4;        // th fp32
static constexpr size_t AB_OFF  = THB_OFF + (size_t)BDIM * ODIM * 4;        // r*h bf16
static constexpr size_t UZV_OFF = AB_OFF  + (size_t)BDIM * ODIM * 2;        // uz fp32
static constexpr size_t URV_OFF = UZV_OFF + (size_t)ODIM * 4;               // ur fp32

extern "C" void kernel_launch(void* const* d_in, const int* in_sizes, int n_in,
                              void* d_out, int out_size, void* d_ws, size_t ws_size,
                              hipStream_t stream) {
  const float* x   = (const float*)d_in[0];
  const float* h   = (const float*)d_in[1];
  const float* Wz  = (const float*)d_in[2];
  const float* bwz = (const float*)d_in[3];
  const float* Uz  = (const float*)d_in[4];
  const float* buz = (const float*)d_in[5];
  const float* Wr  = (const float*)d_in[6];
  const float* bwr = (const float*)d_in[7];
  const float* Ur  = (const float*)d_in[8];
  const float* bur = (const float*)d_in[9];
  const float* Wh  = (const float*)d_in[10];
  const float* bwh = (const float*)d_in[11];
  const float* Uh  = (const float*)d_in[12];
  const float* buh = (const float*)d_in[13];
  const float* b_z = (const float*)d_in[14];
  const float* b_r = (const float*)d_in[15];
  const float* b_h = (const float*)d_in[16];
  float* out = (float*)d_out;

  char* ws = (char*)d_ws;
  __bf16* xb  = (__bf16*)(ws + XB_OFF);
  __bf16* wzb = (__bf16*)(ws + WZB_OFF);
  __bf16* wrb = (__bf16*)(ws + WRB_OFF);
  __bf16* whb = (__bf16*)(ws + WHB_OFF);
  __bf16* uhb = (__bf16*)(ws + UHB_OFF);
  float*  zbuf  = (float*)(ws + ZB_OFF);
  float*  thbuf = (float*)(ws + THB_OFF);
  __bf16* abuf  = (__bf16*)(ws + AB_OFF);
  float*  uzv = (float*)(ws + UZV_OFF);
  float*  urv = (float*)(ws + URV_OFF);

  // K0: converts
  {
    const int nx4 = BDIM * IDIM / 4;
    const int nw4 = ODIM * IDIM / 4;
    gru_cvt_bf16<<<(nx4 + 255) / 256, 256, 0, stream>>>((const v4f*)x,  (v4bf*)xb,  nx4);
    gru_cvt_bf16<<<(nw4 + 255) / 256, 256, 0, stream>>>((const v4f*)Wz, (v4bf*)wzb, nw4);
    gru_cvt_bf16<<<(nw4 + 255) / 256, 256, 0, stream>>>((const v4f*)Wr, (v4bf*)wrb, nw4);
    gru_cvt_bf16<<<(nw4 + 255) / 256, 256, 0, stream>>>((const v4f*)Wh, (v4bf*)whb, nw4);
    gru_cvt_bf16<<<(nw4 + 255) / 256, 256, 0, stream>>>((const v4f*)Uh, (v4bf*)uhb, nw4);
  }

  // K1: h@Uz.T + buz, h@Ur.T + bur (fp32 GEMVs over the single h row)
  gru_gemv<<<(2 * ODIM) / 256, 256, 0, stream>>>(h, Uz, buz, Ur, bur, uzv, urv);

  // K2: fused 3-gate GEMM + sigmoid epilogue
  {
    dim3 grid(ODIM / 64, BDIM / 64);
    gru_gates_wmma<<<grid, 256, 0, stream>>>(xb, wzb, wrb, whb,
                                             bwz, bwr, bwh, uzv, urv, buh,
                                             b_z, b_r, b_h, h,
                                             zbuf, thbuf, abuf);
  }

  // K3: (r*h)@Uh.T + tanh/blend epilogue
  {
    dim3 grid(ODIM / 128, BDIM / 64);
    gru_out_wmma<<<grid, 256, 0, stream>>>(abuf, uhb, thbuf, zbuf, h, out);
  }
}
